// BatchHoppy_81346680586350
// MI455X (gfx1250) — matrix operand order repaired and checked by reading, not compiled, not run on hardware
//
#include <hip/hip_runtime.h>
#include <hip/hip_bf16.h>
#include <math.h>

// ---------------------------------------------------------------------------
// Problem constants (from the reference)
// ---------------------------------------------------------------------------
constexpr int B = 16, N = 2048, F = 2048, E = 128, K = 10, R = 2, H = 2;
constexpr float NEG_BIG = -1e30f;   // log(0) stand-in; __expf -> 0

typedef __attribute__((ext_vector_type(16))) __bf16 v16bf;
typedef __attribute__((ext_vector_type(8)))  float  v8f;

// ---------------------------------------------------------------------------
// Workspace layout (float units)
// ---------------------------------------------------------------------------
constexpr size_t WS_H     = 0;                            // h[R*H][B][E]
constexpr size_t WS_ENORM = WS_H     + (size_t)R*H*B*E;   // ent norms [B*N]
constexpr size_t WS_FNORM = WS_ENORM + (size_t)B*N;       // fact norms [3][B*F]
constexpr size_t WS_LKR   = WS_FNORM + (size_t)3*B*F;     // log kr  - 0.5*|fY|^2  [R][B*F]
constexpr size_t WS_LKRR  = WS_LKR   + (size_t)R*B*F;     // log krr - 0.5*|fZ|^2  [R][B*F]
constexpr size_t WS_SP    = WS_LKRR  + (size_t)R*B*F;     // sp  [B*N]
constexpr size_t WS_ZIDX  = WS_SP    + (size_t)B*N;       // int  [B*K]
constexpr size_t WS_ZSC   = WS_ZIDX  + (size_t)B*K;       // [B*K]
constexpr size_t WS_S0    = WS_ZSC   + (size_t)B*K;       // [B]
constexpr size_t WS_RULE  = WS_S0    + (size_t)B;         // [R][B][K]

// ---------------------------------------------------------------------------
// 1) Row squared-norms: one wave per 128-float row
// ---------------------------------------------------------------------------
__global__ void norms_kernel(const float* __restrict__ src,
                             float* __restrict__ dst, int rows) {
  int gtid = blockIdx.x * blockDim.x + threadIdx.x;
  int row  = gtid >> 5;
  int lane = threadIdx.x & 31;
  if (row >= rows) return;
  const float4 v = *(const float4*)(src + (size_t)row * E + lane * 4);
  float s = v.x * v.x + v.y * v.y + v.z * v.z + v.w * v.w;
  #pragma unroll
  for (int off = 16; off; off >>= 1) s += __shfl_xor(s, off, 32);
  if (lane == 0) dst[row] = s;
}

// ---------------------------------------------------------------------------
// 2) Hop relations h[rj][b][:] = rel[b] @ W[rj] + bias[rj]
//    grid (R*H, B), block 128
// ---------------------------------------------------------------------------
__global__ void hop_kernel(const float* __restrict__ rel,
                           const float* __restrict__ Wm,
                           const float* __restrict__ bias,
                           float* __restrict__ h) {
  int rj = blockIdx.x, b = blockIdx.y, t = threadIdx.x;
  __shared__ float rs[E];
  rs[t] = rel[b * E + t];
  __syncthreads();
  const float* Wp = Wm + (size_t)rj * E * E;
  float acc = bias[rj * E + t];
  for (int e = 0; e < E; ++e) acc = fmaf(rs[e], Wp[e * E + t], acc);
  h[((size_t)rj * B + b) * E + t] = acc;
}

// ---------------------------------------------------------------------------
// 3) scores_0[b] = exp( max_{f<nb} -0.5*(d(rel,frel)+d(a1,fa1)+d(a2,fa2)) )
//    grid B, block 256  (log-domain max; single exp at the end)
// ---------------------------------------------------------------------------
__global__ void scores0_kernel(const float* __restrict__ rel,
                               const float* __restrict__ a1,
                               const float* __restrict__ a2,
                               const float* __restrict__ frel,
                               const float* __restrict__ fa1,
                               const float* __restrict__ fa2,
                               const int* __restrict__ nbf,
                               float* __restrict__ s0) {
  __shared__ float rs[E], a1s[E], a2s[E];
  __shared__ float red[256];
  int b = blockIdx.x, t = threadIdx.x;
  if (t < E) { rs[t] = rel[b*E+t]; a1s[t] = a1[b*E+t]; a2s[t] = a2[b*E+t]; }
  __syncthreads();
  int nb = nbf[b];
  float m = NEG_BIG;
  for (int f = t; f < F; f += 256) {
    if (f >= nb) continue;
    const float* p1 = frel + ((size_t)b*F + f) * E;
    const float* p2 = fa1  + ((size_t)b*F + f) * E;
    const float* p3 = fa2  + ((size_t)b*F + f) * E;
    float d = 0.f;
    for (int e = 0; e < E; ++e) {
      float t1 = rs[e]  - p1[e];
      float t2 = a1s[e] - p2[e];
      float t3 = a2s[e] - p3[e];
      d += t1*t1 + t2*t2 + t3*t3;
    }
    m = fmaxf(m, -0.5f * d);
  }
  red[t] = m; __syncthreads();
  for (int s = 128; s > 0; s >>= 1) {
    if (t < s) red[t] = fmaxf(red[t], red[t + s]);
    __syncthreads();
  }
  if (t == 0) s0[b] = __expf(red[0]);
}

// ---------------------------------------------------------------------------
// 4) Log-domain fact coefficients per (r,b,f):
//    lkr  = -0.5*(||h1-frel||^2 + ||arg1-fA||^2) - 0.5*||fY||^2   (hop-1)
//    lkrr = -0.5*(||h2-frel||^2 + ||arg2-fB||^2) - 0.5*||fZ||^2   (hop-2)
//    fA,fY,fB,fZ per rule:  r0: A=fa1 Y=fa2  B=fa2 Z=fa1
//                           r1: A=fa2 Y=fa1  B=fa1 Z=fa2
//    masked facts -> NEG_BIG.   grid (B, R), block 256
// ---------------------------------------------------------------------------
__global__ void krkrr_kernel(const float* __restrict__ h,
                             const float* __restrict__ a1,
                             const float* __restrict__ a2,
                             const float* __restrict__ frel,
                             const float* __restrict__ fa1,
                             const float* __restrict__ fa2,
                             const float* __restrict__ fn2,   // [3][B*F]: rel, a1, a2
                             const int* __restrict__ nbf,
                             float* __restrict__ lkr,
                             float* __restrict__ lkrr) {
  int b = blockIdx.x, r = blockIdx.y, t = threadIdx.x;
  __shared__ float h1s[E], h2s[E], a1s[E], a2s[E];
  if (t < E) {
    h1s[t] = h[(((size_t)r*H + 0)*B + b)*E + t];
    h2s[t] = h[(((size_t)r*H + 1)*B + b)*E + t];
    a1s[t] = a1[b*E + t];
    a2s[t] = a2[b*E + t];
  }
  __syncthreads();
  int nb = nbf[b];
  const float* fA   = (r == 0) ? fa1 : fa2;
  const float* fB   = (r == 0) ? fa2 : fa1;
  const float* fn2Y = fn2 + (size_t)((r == 0) ? 2 : 1) * B * F;  // kE source norms
  const float* fn2Z = fn2 + (size_t)((r == 0) ? 1 : 2) * B * F;  // z-branch norms
  for (int f = t; f < F; f += 256) {
    const float* pr = frel + ((size_t)b*F + f) * E;
    const float* pa = fA   + ((size_t)b*F + f) * E;
    const float* pb = fB   + ((size_t)b*F + f) * E;
    float d1 = 0.f, d2 = 0.f, d3 = 0.f, d4 = 0.f;
    for (int e = 0; e < E; ++e) {
      float xr = pr[e];
      float u1 = h1s[e] - xr;    d1 += u1*u1;
      float u2 = h2s[e] - xr;    d2 += u2*u2;
      float u3 = a1s[e] - pa[e]; d3 += u3*u3;
      float u4 = a2s[e] - pb[e]; d4 += u4*u4;
    }
    bool valid = (f < nb);
    size_t o = ((size_t)r*B + b)*F + f;
    lkr [o] = valid ? (-0.5f*(d1 + d3) - 0.5f*fn2Y[(size_t)b*F + f]) : NEG_BIG;
    lkrr[o] = valid ? (-0.5f*(d2 + d4) - 0.5f*fn2Z[(size_t)b*F + f]) : NEG_BIG;
  }
}

// ---------------------------------------------------------------------------
// 5) WMMA fused kernel (log domain):
//    sp[b,n] = ent_mask * exp( max_f [ lkr[f] + dot(ent_n,fact_f) ] - 0.5*|n|^2 )
//    dot via bf16 WMMA (f32 acc). Inner epilogue = add+max only; one exp/row.
//    grid (B, N/128), block 256 (8 waves), wave handles 16 rows of n.
// ---------------------------------------------------------------------------
__global__ void sp_wmma_kernel(const float* __restrict__ ent,
                               const float* __restrict__ facts,
                               const float* __restrict__ lkr,    // [B*F], pre-offset by r
                               const float* __restrict__ en2,    // [B*N]
                               const int* __restrict__ nbe,
                               float* __restrict__ sp) {
  const int lane = threadIdx.x & 31;
  const int wave = threadIdx.x >> 5;
  const int b    = blockIdx.x;
  const int n0   = blockIdx.y * 128 + wave * 16;
  const int half = lane >> 4;     // 0: lanes 0-15, 1: lanes 16-31
  const int l15  = lane & 15;

  // --- A tile: 16 rows of entity embeddings (bf16), held across the f loop.
  // 16-bit A 16x32 layout: lanes<16 hold K {0..7, 16..23}; lanes>=16 {8..15, 24..31}.
  const float* entRow = ent + ((size_t)b * N + (n0 + l15)) * E;
  v16bf a[4];
  #pragma unroll
  for (int kk = 0; kk < 4; ++kk) {
    const int base = kk * 32 + half * 8;
    #pragma unroll
    for (int i = 0; i < 8; ++i) {
      a[kk][i]     = (__bf16)entRow[base + i];
      a[kk][i + 8] = (__bf16)entRow[base + 16 + i];
    }
  }

  const int nb_row = n0 + half * 8;   // C layout: VGPR r -> M=r (half 0) / r+8
  float vmax[8];
  #pragma unroll
  for (int r = 0; r < 8; ++r) vmax[r] = NEG_BIG;

  const float* factB = facts + (size_t)b * F * E;

  for (int f0 = 0; f0 < F; f0 += 16) {
    // --- B tile: 16 fact rows. 16-bit B 32x16 layout: lane holds column
    // f = l15, K {0..7, 16..23} for lanes<16 and {8..15, 24..31} for lanes>=16.
    const float* fr = factB + (size_t)(f0 + l15) * E;
    v8f acc = {0.f, 0.f, 0.f, 0.f, 0.f, 0.f, 0.f, 0.f};
    #pragma unroll
    for (int kk = 0; kk < 4; ++kk) {
      const int base = kk * 32 + half * 8;
      v16bf bt;
      #pragma unroll
      for (int i = 0; i < 8; ++i) {
        bt[i]     = (__bf16)fr[base + i];
        bt[i + 8] = (__bf16)fr[base + 16 + i];
      }
      acc = __builtin_amdgcn_wmma_f32_16x16x32_bf16(
          /*neg_a=*/false, a[kk], /*neg_b=*/false, bt,
          /*c_mod=*/(short)0, acc, /*reuse_a=*/false, /*reuse_b=*/false);
    }
    // Log-domain epilogue: running max of (dot + lkr[f]).  add+max only.
    const float lkrf = lkr[(size_t)b * F + f0 + l15];
    #pragma unroll
    for (int r = 0; r < 8; ++r)
      vmax[r] = fmaxf(vmax[r], acc[r] + lkrf);
  }

  // Add row-norm term (constant per n-row), reduce across the 16 lanes
  // sharing each row (xor 1..8 stays within each half), exp once, mask.
  const int nent = nbe[b];
  #pragma unroll
  for (int r = 0; r < 8; ++r) {
    float v = vmax[r] - 0.5f * en2[(size_t)b * N + nb_row + r];
    #pragma unroll
    for (int off = 1; off <= 8; off <<= 1) v = fmaxf(v, __shfl_xor(v, off, 32));
    if (l15 == 0) {
      int n = nb_row + r;
      sp[(size_t)b * N + n] = (n < nent) ? __expf(v) : 0.f;
    }
  }
}

// ---------------------------------------------------------------------------
// 6) top-K per batch (iterative argmax, K=10 small). grid B, block 256.
//    Destroys sp (scratch; fully rewritten per r pass).
// ---------------------------------------------------------------------------
__global__ void topk_kernel(float* __restrict__ sp,
                            int* __restrict__ zidx,
                            float* __restrict__ zsc) {
  __shared__ float sv[256];
  __shared__ int   si[256];
  int b = blockIdx.x, t = threadIdx.x;
  float* row = sp + (size_t)b * N;
  for (int k = 0; k < K; ++k) {
    float bv = -1.f; int bi = N;
    for (int n = t; n < N; n += 256) {
      float v = row[n];
      if (v > bv) { bv = v; bi = n; }
    }
    sv[t] = bv; si[t] = bi; __syncthreads();
    for (int s = 128; s > 0; s >>= 1) {
      if (t < s) {
        bool take = (sv[t+s] > sv[t]) || (sv[t+s] == sv[t] && si[t+s] < si[t]);
        if (take) { sv[t] = sv[t+s]; si[t] = si[t+s]; }
      }
      __syncthreads();
    }
    if (t == 0) {
      zidx[b*K + k] = si[0];
      zsc [b*K + k] = sv[0];
      row[si[0]] = -1.f;
    }
    __syncthreads();
  }
}

// ---------------------------------------------------------------------------
// 7) Final hop (log domain):
//    rule[r][b][k] = min(z_score, exp( max_f [lkrr[f] + dot(z_k,f)] - 0.5|z|^2 ))
//    grid (K, B), block 256.
// ---------------------------------------------------------------------------
__global__ void finalhop_kernel(const float* __restrict__ ent,
                                const float* __restrict__ factZ,
                                const float* __restrict__ en2,   // [B*N]
                                const float* __restrict__ lkrr,  // [B*F], pre-offset by r
                                const int* __restrict__ zidx,
                                const float* __restrict__ zsc,
                                float* __restrict__ rule) {      // pre-offset: [B*K]
  __shared__ float zrow[E];
  __shared__ float red[256];
  int k = blockIdx.x, b = blockIdx.y, t = threadIdx.x;
  int zi = zidx[b*K + k];
  if (t < E) zrow[t] = ent[((size_t)b * N + zi) * E + t];
  __syncthreads();
  float m = NEG_BIG;
  for (int f = t; f < F; f += 256) {
    const float* fr = factZ + ((size_t)b * F + f) * E;
    float dot = 0.f;
    for (int e = 0; e < E; e += 4) {
      dot += zrow[e]*fr[e] + zrow[e+1]*fr[e+1]
           + zrow[e+2]*fr[e+2] + zrow[e+3]*fr[e+3];
    }
    m = fmaxf(m, dot + lkrr[(size_t)b*F + f]);
  }
  red[t] = m; __syncthreads();
  for (int s = 128; s > 0; s >>= 1) {
    if (t < s) red[t] = fmaxf(red[t], red[t + s]);
    __syncthreads();
  }
  if (t == 0) {
    float s2 = __expf(red[0] - 0.5f * en2[(size_t)b * N + zi]);
    rule[b*K + k] = fminf(s2, zsc[b*K + k]);
  }
}

// ---------------------------------------------------------------------------
// 8) Combine: out[b] = max(s0[b], max_{r,k} rule[r][b][k])
// ---------------------------------------------------------------------------
__global__ void combine_kernel(const float* __restrict__ s0,
                               const float* __restrict__ rule,
                               float* __restrict__ out) {
  int b = threadIdx.x;
  if (b >= B) return;
  float m = s0[b];
  for (int r = 0; r < R; ++r)
    for (int k = 0; k < K; ++k)
      m = fmaxf(m, rule[((size_t)r * B + b) * K + k]);
  out[b] = m;
}

// ---------------------------------------------------------------------------
// Host launcher
// ---------------------------------------------------------------------------
extern "C" void kernel_launch(void* const* d_in, const int* in_sizes, int n_in,
                              void* d_out, int out_size, void* d_ws, size_t ws_size,
                              hipStream_t stream) {
  const float* rel   = (const float*)d_in[0];
  const float* arg1  = (const float*)d_in[1];
  const float* arg2  = (const float*)d_in[2];
  const float* frel  = (const float*)d_in[3];
  const float* fa1   = (const float*)d_in[4];
  const float* fa2   = (const float*)d_in[5];
  const float* ent   = (const float*)d_in[6];
  const float* hopW  = (const float*)d_in[7];
  const float* hopb  = (const float*)d_in[8];
  const int*   nbf   = (const int*)d_in[9];
  const int*   nbe   = (const int*)d_in[10];

  float* ws    = (float*)d_ws;
  float* h     = ws + WS_H;
  float* en2   = ws + WS_ENORM;
  float* fn2   = ws + WS_FNORM;   // [0]=frel [1]=fa1 [2]=fa2
  float* lkr   = ws + WS_LKR;
  float* lkrr  = ws + WS_LKRR;
  float* sp    = ws + WS_SP;
  int*   zidx  = (int*)(ws + WS_ZIDX);
  float* zsc   = ws + WS_ZSC;
  float* s0    = ws + WS_S0;
  float* rule  = ws + WS_RULE;
  float* out   = (float*)d_out;

  // 1) row norms (one wave per row; 8 rows per 256-thread block)
  {
    int rowsE = B * N, rowsF = B * F;
    norms_kernel<<<dim3((rowsE + 7) / 8), 256, 0, stream>>>(ent,  en2,                  rowsE);
    norms_kernel<<<dim3((rowsF + 7) / 8), 256, 0, stream>>>(frel, fn2 + 0*(size_t)B*F,  rowsF);
    norms_kernel<<<dim3((rowsF + 7) / 8), 256, 0, stream>>>(fa1,  fn2 + 1*(size_t)B*F,  rowsF);
    norms_kernel<<<dim3((rowsF + 7) / 8), 256, 0, stream>>>(fa2,  fn2 + 2*(size_t)B*F,  rowsF);
  }

  // 2) hop relations
  hop_kernel<<<dim3(R * H, B), E, 0, stream>>>(rel, hopW, hopb, h);

  // 3) scores_0
  scores0_kernel<<<dim3(B), 256, 0, stream>>>(rel, arg1, arg2, frel, fa1, fa2, nbf, s0);

  // 4) log-domain kr / krr for both rules (fact norms folded in)
  krkrr_kernel<<<dim3(B, R), 256, 0, stream>>>(h, arg1, arg2, frel, fa1, fa2,
                                               fn2, nbf, lkr, lkrr);

  // 5..7) per-rule: WMMA sp -> top-k -> final hop
  for (int r = 0; r < R; ++r) {
    const float* factY = (r == 0) ? fa2 : fa1;   // kE source (hop 1)
    const float* factZ = (r == 0) ? fa1 : fa2;   // z-branch source (hop 2)

    sp_wmma_kernel<<<dim3(B, N / 128), 256, 0, stream>>>(
        ent, factY, lkr + (size_t)r * B * F, en2, nbe, sp);

    topk_kernel<<<dim3(B), 256, 0, stream>>>(sp, zidx, zsc);

    finalhop_kernel<<<dim3(K, B), 256, 0, stream>>>(
        ent, factZ, en2, lkrr + (size_t)r * B * F, zidx, zsc,
        rule + (size_t)r * B * K);
  }

  // 8) combine
  combine_kernel<<<dim3(1), 32, 0, stream>>>(s0, rule, out);
}
